// IBC_33990371181557
// MI455X (gfx1250) — matrix operand order repaired and checked by reading, not compiled, not run on hardware
//
#include <hip/hip_runtime.h>
#include <hip/hip_bf16.h>

typedef float    v2f   __attribute__((ext_vector_type(2)));
typedef float    v8f   __attribute__((ext_vector_type(8)));
typedef __bf16   v16bf __attribute__((ext_vector_type(16)));
typedef unsigned u32x4 __attribute__((ext_vector_type(4)));
typedef int      i32x4 __attribute__((ext_vector_type(4)));
typedef int      i32x8 __attribute__((ext_vector_type(8)));

#define DIMC   192
#define HEADS_ 8
#define NB_    4
#define BSZ    2
#define HIMG   256
#define WIMG   256
#define HW_    (HIMG*WIMG)
#define CHPH   24                 // DIMC / HEADS
#define HS_    64                 // HIMG / NB
#define TKN    4096               // HS_*HS_
#define RR     384                // CHPH*NB*NB
#define BH     (BSZ*HEADS_)
static const long NTOT = (long)BSZ * DIMC * HW_;   // 25,165,824 elems

// ---------------------------------------------------------------------------
// TDM: DMA a 2-D tile (rows x cols fp32, row stride in elems) from global
// memory into LDS. D# packing per CDNA5 ISA.
// Issued by one wave; completion via TENSORcnt.
// ---------------------------------------------------------------------------
__device__ __forceinline__ void tdm_load_tile_f32(const float* gsrc,
                                                  void* lds_ptr,
                                                  unsigned rows, unsigned cols,
                                                  unsigned long row_stride) {
  unsigned long long ga = (unsigned long long)(uintptr_t)gsrc;
  unsigned lds_off = (unsigned)(uintptr_t)lds_ptr;   // LDS aperture: low 32 bits

  u32x4 g0;
  g0[0] = 1u;                                        // count=1, user mode
  g0[1] = lds_off;                                   // lds_addr (bytes)
  g0[2] = (unsigned)(ga & 0xFFFFFFFFull);            // global_addr[31:0]
  g0[3] = (unsigned)((ga >> 32) & 0x1FFFFFFull)      // global_addr[56:32]
        | (2u << 30);                                // type = 2 ("image")

  i32x8 g1;
  g1[0] = (int)(2u << 16);                           // data_size = 2 -> 4 bytes
  g1[1] = (int)((cols & 0xFFFFu) << 16);             // tensor_dim0[15:0]
  g1[2] = (int)(((cols >> 16) & 0xFFFFu)             // tensor_dim0[31:16]
        |       ((rows & 0xFFFFu) << 16));           // tensor_dim1[15:0]
  g1[3] = (int)(((rows >> 16) & 0xFFFFu)             // tensor_dim1[31:16]
        |       ((cols & 0xFFFFu) << 16));           // tile_dim0
  g1[4] = (int)(rows & 0xFFFFu);                     // tile_dim1 (tile_dim2=0)
  g1[5] = (int)(row_stride & 0xFFFFFFFFull);         // tensor_dim0_stride[31:0]
  g1[6] = (int)((row_stride >> 32) & 0xFFFFull);     // tensor_dim0_stride[47:32]
  g1[7] = 0;                                         // tensor_dim1_stride (2D)

  i32x4 gz = {0, 0, 0, 0};
#if defined(__clang_major__) && (__clang_major__ >= 23)
  i32x8 gz8 = {0, 0, 0, 0, 0, 0, 0, 0};
  __builtin_amdgcn_tensor_load_to_lds(g0, g1, gz, gz, gz8, 0);
#else
  __builtin_amdgcn_tensor_load_to_lds(g0, g1, gz, gz, 0);
#endif
}

// ---------------------------------------------------------------------------
// conv1x1 as GEMM via V_WMMA_F32_16X16X4_F32.
// out[b,o,s] = sum_c w[o,c]*in[b,c,s] + bias[o]
// block = 384 threads = 12 waves; wave w owns o-tile w (12*16 = 192 channels);
// block covers 16 spatial positions. Input tile (192x16 f32, 12KB) is staged
// into LDS once per block by wave 0 via the Tensor Data Mover.
// ---------------------------------------------------------------------------
__global__ __launch_bounds__(384)
void k_conv1x1(const float* __restrict__ in, const float* __restrict__ w,
               const float* __restrict__ bias, float* __restrict__ out) {
  __shared__ float tile[DIMC * 16];                  // [channel][spatial]

  const int wave = threadIdx.x >> 5;
  const int lane = threadIdx.x & 31;
  const int hi   = lane >> 4;     // 0: lanes 0-15, 1: lanes 16-31
  const int ln   = lane & 15;
  const int o0   = wave * 16;

  long p0 = (long)blockIdx.x * 16;          // flat over B*HW
  long b  = p0 / HW_;
  long s  = p0 - b * HW_;
  const float* inb = in + b * DIMC * (long)HW_ + s;

  if (threadIdx.x < 32) {                   // wave 0: one DMA for the block
    tdm_load_tile_f32(inb, tile, DIMC, 16, (unsigned long)HW_);
    __builtin_amdgcn_s_wait_tensorcnt(0);
  }
  __syncthreads();

  v8f acc = {};
  for (int k0 = 0; k0 < DIMC; k0 += 4) {
    const int ka = k0 + hi * 2;
    // A fragment: 16x4 f32. lanes0-15 hold K=0,1; lanes16-31 hold K=2,3; M=ln
    v2f a;
    a.x = w[(o0 + ln) * DIMC + ka];
    a.y = w[(o0 + ln) * DIMC + ka + 1];
    // B fragment: 4x16 f32 from LDS. VGPR0: K=ka, VGPR1: K=ka+1; N=ln
    v2f bfr;
    bfr.x = tile[ka * 16 + ln];
    bfr.y = tile[(ka + 1) * 16 + ln];
    acc = __builtin_amdgcn_wmma_f32_16x16x4_f32(false, a, false, bfr,
                                                (short)0, acc, false, false);
  }
  // D layout: VGPR v -> M = o0 + v + hi*8, N = ln
  #pragma unroll
  for (int v = 0; v < 8; ++v) {
    int o = o0 + v + hi * 8;
    out[(b * DIMC + o) * (long)HW_ + s + ln] = acc[v] + bias[o];
  }
}

// ---------------------------------------------------------------------------
// depthwise 3x3, groups = C, SAME zero padding
// ---------------------------------------------------------------------------
__global__ __launch_bounds__(256)
void k_dw3(const float* __restrict__ in, const float* __restrict__ w,
           float* __restrict__ out) {
  long idx = (long)blockIdx.x * 256 + threadIdx.x;
  if (idx >= NTOT) return;
  int  x  = (int)(idx % WIMG);
  long r  = idx / WIMG;
  int  y  = (int)(r % HIMG);
  long bc = r / HIMG;
  int  ch = (int)(bc % DIMC);
  const float* ip = in + bc * HW_;
  const float* wp = w + ch * 9;
  float acc = 0.f;
  #pragma unroll
  for (int dy = -1; dy <= 1; ++dy) {
    int yy = y + dy;
    if (yy < 0 || yy >= HIMG) continue;
    #pragma unroll
    for (int dx = -1; dx <= 1; ++dx) {
      int xx = x + dx;
      if (xx < 0 || xx >= WIMG) continue;
      acc += ip[yy * WIMG + xx] * wp[(dy + 1) * 3 + (dx + 1)];
    }
  }
  out[idx] = acc;
}

// ---------------------------------------------------------------------------
// MDC: 4-way grouped dwconv (3/5/7/9), concat, gated by sigmoid(x)
// out channel ch: branch j = ch/48, group g = ch%48, inputs g*4 .. g*4+3
// ---------------------------------------------------------------------------
__global__ __launch_bounds__(256)
void k_mdc(const float* __restrict__ in,
           const float* __restrict__ w3, const float* __restrict__ w5,
           const float* __restrict__ w7, const float* __restrict__ w9,
           float* __restrict__ out) {
  long idx = (long)blockIdx.x * 256 + threadIdx.x;
  if (idx >= NTOT) return;
  int  x  = (int)(idx % WIMG);
  long r  = idx / WIMG;
  int  y  = (int)(r % HIMG);
  long bc = r / HIMG;
  int  ch = (int)(bc % DIMC);
  long b  = bc / DIMC;
  int  j  = ch / 48;
  int  g  = ch % 48;
  const float* wp; int ks;
  if      (j == 0) { wp = w3; ks = 3; }
  else if (j == 1) { wp = w5; ks = 5; }
  else if (j == 2) { wp = w7; ks = 7; }
  else             { wp = w9; ks = 9; }
  const int half = ks >> 1;
  float acc = 0.f;
  for (int ci = 0; ci < 4; ++ci) {
    const float* ip = in + (b * DIMC + g * 4 + ci) * (long)HW_;
    const float* wc = wp + (g * 4 + ci) * ks * ks;
    for (int dy = -half; dy <= half; ++dy) {
      int yy = y + dy;
      if (yy < 0 || yy >= HIMG) continue;
      for (int dx = -half; dx <= half; ++dx) {
        int xx = x + dx;
        if (xx < 0 || xx >= WIMG) continue;
        acc += ip[yy * WIMG + xx] * wc[(dy + half) * ks + (dx + half)];
      }
    }
  }
  float xv = in[(b * DIMC + ch) * (long)HW_ + y * WIMG + x];
  out[idx] = acc / (1.f + __expf(-xv));
}

// ---------------------------------------------------------------------------
// head reshape (+ optional L2-norm over 4096 tokens), write bf16
// ---------------------------------------------------------------------------
__global__ __launch_bounds__(256)
void k_heads(const float* __restrict__ in, __bf16* __restrict__ outh,
             int do_norm) {
  const int row = blockIdx.x;                   // 0 .. BH*RR-1
  const int r   = row % RR;
  const int bh  = row / RR;
  const int h   = bh % HEADS_;
  const int b   = bh / HEADS_;
  const int cc  = r / 16;
  const int h1  = (r >> 2) & 3;
  const int w1  = r & 3;
  const int ch  = h * CHPH + cc;
  const float* ip = in + ((long)(b * DIMC + ch)) * HW_
                      + (long)(h1 * HS_) * WIMG + w1 * HS_;
  __shared__ float red[256];
  float vals[16];
  float ss = 0.f;
  #pragma unroll
  for (int i = 0; i < 16; ++i) {
    int t  = i * 256 + threadIdx.x;
    int hs = t >> 6, ws = t & 63;
    float v = ip[hs * WIMG + ws];
    vals[i] = v;
    ss += v * v;
  }
  red[threadIdx.x] = ss;
  __syncthreads();
  for (int st = 128; st > 0; st >>= 1) {
    if (threadIdx.x < st) red[threadIdx.x] += red[threadIdx.x + st];
    __syncthreads();
  }
  float scale = 1.f;
  if (do_norm) scale = 1.f / fmaxf(sqrtf(red[0]), 1e-12f);
  __bf16* op = outh + (long)row * TKN;
  #pragma unroll
  for (int i = 0; i < 16; ++i) {
    int t = i * 256 + threadIdx.x;
    op[t] = (__bf16)(vals[i] * scale);
  }
}

// ---------------------------------------------------------------------------
// S[bh, r1, r2] = temp[h] * sum_t Q[bh,r1,t] * K[bh,r2,t]
// WMMA bf16 16x16x32. grid = (24 r1-tiles, 3, BH), block = 256 (8 r2 waves)
// ---------------------------------------------------------------------------
__global__ __launch_bounds__(256)
void k_gemm_qk(const __bf16* __restrict__ Qh, const __bf16* __restrict__ Kh,
               const float* __restrict__ temp, float* __restrict__ S) {
  const int wave = threadIdx.x >> 5;
  const int lane = threadIdx.x & 31;
  const int hi   = lane >> 4;
  const int ln   = lane & 15;
  const int bh   = blockIdx.z;
  const int r1b  = blockIdx.x * 16;
  const int r2b  = (blockIdx.y * 8 + wave) * 16;

  const __bf16* qrow = Qh + ((long)bh * RR + r1b + ln) * TKN;
  const __bf16* krow = Kh + ((long)bh * RR + r2b + ln) * TKN;

  v8f acc = {};
  for (int k0 = 0; k0 < TKN; k0 += 32) {
    __builtin_prefetch(qrow + k0 + 256, 0, 1);
    __builtin_prefetch(krow + k0 + 256, 0, 1);
    v16bf a, bb;
    #pragma unroll
    for (int i = 0; i < 16; ++i) {
      a[i]     = qrow[k0 + hi * 16 + i];
    }
    #pragma unroll
    for (int i = 0; i < 16; ++i) {
      bb[i] = krow[k0 + hi * 16 + i];
    }
    acc = __builtin_amdgcn_wmma_f32_16x16x32_bf16(false, a, false, bb,
                                                  (short)0, acc, false, false);
  }
  const float ts = temp[bh % HEADS_];
  #pragma unroll
  for (int v = 0; v < 8; ++v) {
    int m = r1b + v + hi * 8;
    S[((long)bh * RR + m) * RR + r2b + ln] = acc[v] * ts;
  }
}

// ---------------------------------------------------------------------------
// row softmax over S (rows of length 384). One block (128 thr) per row.
// ---------------------------------------------------------------------------
__global__ __launch_bounds__(128)
void k_softmax(float* __restrict__ S) {
  const long row = blockIdx.x;
  float* p = S + row * RR;
  __shared__ float red[128];
  float v[3];
  float mx = -3.4e38f;
  #pragma unroll
  for (int i = 0; i < 3; ++i) { v[i] = p[i * 128 + threadIdx.x]; mx = fmaxf(mx, v[i]); }
  red[threadIdx.x] = mx;
  __syncthreads();
  for (int st = 64; st > 0; st >>= 1) {
    if (threadIdx.x < st) red[threadIdx.x] = fmaxf(red[threadIdx.x], red[threadIdx.x + st]);
    __syncthreads();
  }
  mx = red[0];
  __syncthreads();
  float sum = 0.f;
  #pragma unroll
  for (int i = 0; i < 3; ++i) { v[i] = __expf(v[i] - mx); sum += v[i]; }
  red[threadIdx.x] = sum;
  __syncthreads();
  for (int st = 64; st > 0; st >>= 1) {
    if (threadIdx.x < st) red[threadIdx.x] += red[threadIdx.x + st];
    __syncthreads();
  }
  float inv = 1.f / red[0];
  #pragma unroll
  for (int i = 0; i < 3; ++i) p[i * 128 + threadIdx.x] = v[i] * inv;
}

// ---------------------------------------------------------------------------
// O[bh, r1, t] = sum_r2 P[bh,r1,r2] * V[bh,r2,t]  (P fp32 -> bf16 on load)
// grid = (24 r1-tiles, 32, BH), block = 256 (8 t waves)
// ---------------------------------------------------------------------------
__global__ __launch_bounds__(256)
void k_gemm_pv(const float* __restrict__ P, const __bf16* __restrict__ Vh,
               float* __restrict__ Oh) {
  const int wave = threadIdx.x >> 5;
  const int lane = threadIdx.x & 31;
  const int hi   = lane >> 4;
  const int ln   = lane & 15;
  const int bh   = blockIdx.z;
  const int r1b  = blockIdx.x * 16;
  const int tb   = (blockIdx.y * 8 + wave) * 16;

  const float* prow = P + ((long)bh * RR + r1b + ln) * RR;

  v8f acc = {};
  for (int k0 = 0; k0 < RR; k0 += 32) {
    v16bf a, bb;
    #pragma unroll
    for (int i = 0; i < 16; ++i) {
      a[i]     = (__bf16)prow[k0 + hi * 16 + i];
    }
    #pragma unroll
    for (int i = 0; i < 16; ++i) {
      int k = k0 + hi * 16 + i;
      bb[i] = Vh[((long)bh * RR + k) * TKN + tb + ln];
    }
    acc = __builtin_amdgcn_wmma_f32_16x16x32_bf16(false, a, false, bb,
                                                  (short)0, acc, false, false);
  }
  #pragma unroll
  for (int v = 0; v < 8; ++v) {
    int m = r1b + v + hi * 8;
    Oh[((long)bh * RR + m) * TKN + tb + ln] = acc[v];
  }
}

// ---------------------------------------------------------------------------
// from_heads: Oh[b,head,r,t] -> F[b,ch,y,x]
// ---------------------------------------------------------------------------
__global__ __launch_bounds__(256)
void k_from_heads(const float* __restrict__ Oh, float* __restrict__ F) {
  long idx = (long)blockIdx.x * 256 + threadIdx.x;
  if (idx >= NTOT) return;
  int  x  = (int)(idx % WIMG);
  long r0 = idx / WIMG;
  int  y  = (int)(r0 % HIMG);
  long bc = r0 / HIMG;
  int  ch = (int)(bc % DIMC);
  long b  = bc / DIMC;
  int  h  = ch / CHPH, cc = ch % CHPH;
  int  h1 = y / HS_, hs = y % HS_;
  int  w1 = x / HS_, ws = x % HS_;
  int  r  = (cc * NB_ + h1) * NB_ + w1;
  int  t  = hs * HS_ + ws;
  F[idx] = Oh[(((long)(b * HEADS_ + h)) * RR + r) * TKN + t];
}

// ---------------------------------------------------------------------------
extern "C" void kernel_launch(void* const* d_in, const int* in_sizes, int n_in,
                              void* d_out, int out_size, void* d_ws, size_t ws_size,
                              hipStream_t stream) {
  const float* xu     = (const float*)d_in[0];
  const float* xd     = (const float*)d_in[1];
  const float* temp   = (const float*)d_in[2];
  const float* q1_w   = (const float*)d_in[3];
  const float* q1_b   = (const float*)d_in[4];
  const float* k1_w   = (const float*)d_in[5];
  const float* k1_b   = (const float*)d_in[6];
  const float* v1_w   = (const float*)d_in[7];
  const float* v1_b   = (const float*)d_in[8];
  const float* mdcq_w = (const float*)d_in[9];
  const float* mdck_w = (const float*)d_in[10];
  const float* mdc1_w = (const float*)d_in[11];
  const float* mdc2_w = (const float*)d_in[12];
  const float* mdc3_w = (const float*)d_in[13];
  const float* mdc4_w = (const float*)d_in[14];
  const float* proj_w = (const float*)d_in[15];
  const float* proj_b = (const float*)d_in[16];

  const long N = NTOT;
  float*  A  = (float*)d_ws;
  float*  Bf = A + N;
  float*  Cf = Bf + N;
  float*  D  = Cf + N;
  __bf16* Qh = (__bf16*)(D + N);
  __bf16* Kh = Qh + N;
  __bf16* Vh = Kh + N;
  float*  S  = (float*)(Vh + N);                 // BH*RR*RR floats
  float*  Oh = A;                                // reuse A after Vh built
  float*  F  = D;                                // reuse D after Qh built
  float*  out = (float*)d_out;

  const int gConv = (int)(N / DIMC / 16);        // B*HW/16 = 8192
  const int gElem = (int)((N + 255) / 256);      // 98304

  // q path
  k_conv1x1<<<gConv, 384, 0, stream>>>(xu, q1_w, q1_b, A);
  k_dw3<<<gElem, 256, 0, stream>>>(A, mdcq_w, D);                    // q1 -> D
  // k path
  k_conv1x1<<<gConv, 384, 0, stream>>>(xd, k1_w, k1_b, A);
  k_dw3<<<gElem, 256, 0, stream>>>(A, mdck_w, Bf);                   // k1 -> B
  // v path
  k_conv1x1<<<gConv, 384, 0, stream>>>(xd, v1_w, v1_b, Cf);
  k_mdc<<<gElem, 256, 0, stream>>>(Cf, mdc1_w, mdc2_w, mdc3_w, mdc4_w, A); // v1 -> A

  // heads + normalize -> bf16
  k_heads<<<BH * RR, 256, 0, stream>>>(D,  Qh, 1);
  k_heads<<<BH * RR, 256, 0, stream>>>(Bf, Kh, 1);
  k_heads<<<BH * RR, 256, 0, stream>>>(A,  Vh, 0);

  // attention
  k_gemm_qk<<<dim3(RR / 16, 3, BH), 256, 0, stream>>>(Qh, Kh, temp, S);
  k_softmax<<<BH * RR, 128, 0, stream>>>(S);
  k_gemm_pv<<<dim3(RR / 16, 32, BH), 256, 0, stream>>>(S, Vh, Oh);

  // merge heads + final projection
  k_from_heads<<<gElem, 256, 0, stream>>>(Oh, F);
  k_conv1x1<<<gConv, 384, 0, stream>>>(F, proj_w, proj_b, out);
}
